// MultiHeadAttention_3092376453581
// MI455X (gfx1250) — matrix operand order repaired and checked by reading, not compiled, not run on hardware
//
#include <hip/hip_runtime.h>
#include <hip/hip_bf16.h>

#define S_LEN  4096
#define DMODEL 512
#define NHEADS 8
#define HDIM   64
#define BATCH  2

typedef __attribute__((ext_vector_type(16))) _Float16 v16h;
typedef __attribute__((ext_vector_type(8)))  _Float16 v8h;
typedef __attribute__((ext_vector_type(8)))  float    v8f;

// ---------------------------------------------------------------------------
// WMMA helper (CDNA5 wave32, V_WMMA_F32_16X16X32_F16)
// ---------------------------------------------------------------------------
__device__ __forceinline__ v8f wmma16(v16h a, v16h b, v8f c) {
  return __builtin_amdgcn_wmma_f32_16x16x32_f16(false, a, false, b,
                                                (short)0, c, false, false);
}

// A fragment (16x32 MxK f16): lane (l, half) holds row m0+l,
// elems 0..15 = K[k0..k0+15], k0 = kk + 16*half (16 contiguous).
__device__ __forceinline__ v16h a_frag_f32(const float* __restrict__ row, int k0) {
  v16h r;
#pragma unroll
  for (int j = 0; j < 8; ++j) {
    r[j]     = (_Float16)row[k0 + j];
    r[j + 8] = (_Float16)row[k0 + 8 + j];
  }
  return r;
}
__device__ __forceinline__ v16h a_frag_f16(const _Float16* row, int k0) {
  v8h lo = *(const v8h*)(row + k0);
  v8h hi = *(const v8h*)(row + k0 + 8);
  v16h r;
#pragma unroll
  for (int j = 0; j < 8; ++j) { r[j] = lo[j]; r[j + 8] = hi[j]; }
  return r;
}
// B fragment (32x16 KxN f16): lane (n=l, half) holds col n, elems = K[k0+j],
// k0 = kk + 16*half (16 contiguous).
__device__ __forceinline__ v16h b_frag_f32(const float* row, int k0) {
  v16h r;
#pragma unroll
  for (int j = 0; j < 16; ++j) r[j] = (_Float16)row[k0 + j];
  return r;
}
__device__ __forceinline__ v16h b_frag_f16(const _Float16* __restrict__ row, int k0) {
  return *(const v16h*)(row + k0);
}

// async copy 16B global->LDS (gfx1250 GLOBAL_LOAD_ASYNC_TO_LDS_B128, ASYNCcnt)
#define ASYNC_CP16(ldsaddr, gaddr, OFF)                                        \
  asm volatile("global_load_async_to_lds_b128 %0, %1, off offset:" #OFF        \
               :: "v"(ldsaddr), "v"(gaddr) : "memory")

// ---------------------------------------------------------------------------
// 1) QKV projection: Y = X @ W^T + b (f32 in, f16 out). 8 waves share one
//    n-tile; the 16x512 W tile is staged once into LDS with async copies.
//    v_transposed==0 -> out[bh][s][d] (Q,K); ==1 -> out[bh][d][s] (V)
// ---------------------------------------------------------------------------
#define WT_STRIDE 516   // 512 + 4 pad floats: lanes hit distinct LDS banks

__global__ __launch_bounds__(256) void qkv_proj_kernel(
    const float* __restrict__ X, const float* __restrict__ W,
    const float* __restrict__ bias, _Float16* __restrict__ out,
    int v_transposed) {
  __shared__ float wtile[16 * WT_STRIDE];            // ~33 KB

  const int t = threadIdx.x;
  const int n0 = blockIdx.x * 16;

  // ---- stage W rows n0..n0+15 (16 x 512 f32) via async LDS DMA -----------
  {
    const int r = t >> 4, seg = t & 15;              // thread: row r, 32 floats
    unsigned int la =
        (unsigned int)(size_t)&wtile[r * WT_STRIDE + seg * 32];
    unsigned long long ga =
        (unsigned long long)(W + (size_t)(n0 + r) * DMODEL + seg * 32);
    ASYNC_CP16(la, ga, 0);
    ASYNC_CP16(la, ga, 16);
    ASYNC_CP16(la, ga, 32);
    ASYNC_CP16(la, ga, 48);
    ASYNC_CP16(la, ga, 64);
    ASYNC_CP16(la, ga, 80);
    ASYNC_CP16(la, ga, 96);
    ASYNC_CP16(la, ga, 112);
    asm volatile("s_wait_asynccnt 0x0" ::: "memory");
  }
  __syncthreads();

  const int w = t >> 5, lane = t & 31, l = lane & 15, half = lane >> 4;
  const int m0 = (blockIdx.y * 8 + w) * 16;
  const float* arow  = X + (size_t)(m0 + l) * DMODEL;
  const float* brow  = &wtile[l * WT_STRIDE];        // B[k,n] = W[n0+l][k]
  v8f acc = {};
#pragma unroll 4
  for (int kk = 0; kk < DMODEL; kk += 32)
    acc = wmma16(a_frag_f32(arow, kk + 16 * half),
                 b_frag_f32(brow, kk + 16 * half), acc);

  const int n = n0 + l, h = n >> 6, d = n & (HDIM - 1);
  const float bv = bias[n];
#pragma unroll
  for (int i = 0; i < 8; ++i) {
    const int m  = m0 + i + 8 * half;
    const int b_ = m >> 12, s = m & (S_LEN - 1);
    const int bh = b_ * NHEADS + h;
    const float y = acc[i] + bv;
    const size_t idx = v_transposed
        ? ((size_t)bh * HDIM + d) * S_LEN + s
        : ((size_t)bh * S_LEN + s) * HDIM + d;
    out[idx] = (_Float16)y;
  }
}

// ---------------------------------------------------------------------------
// 2) Fused attention: one 8-wave workgroup per (bh, 16 query rows).
//    QK^T -> LDS strip -> softmax stats -> single prob write -> PV from LDS.
// ---------------------------------------------------------------------------
#define SC_STRIDE 4104  // 4096 + 8 f16 pad (16B) -> conflict-free ds_load_b128

__global__ __launch_bounds__(256) void attn_fused_kernel(
    const _Float16* __restrict__ Qh, const _Float16* __restrict__ Kh,
    const _Float16* __restrict__ Vt, float* __restrict__ attn,
    _Float16* __restrict__ ctx) {
  __shared__ _Float16 sc[16 * SC_STRIDE];            // 128.25 KB score strip
  __shared__ float    scr8[8][16][16];               // wave scratch (8 KB)
  __shared__ float    red[16][16];
  __shared__ float    rowmax[16], rowinv[16];

  const int t = threadIdx.x;
  const int w = t >> 5, lane = t & 31, l = lane & 15, half = lane >> 4;
  const int bh = blockIdx.y;
  const int m0 = blockIdx.x * 16;

  // ---- phase 1: scores = QK^T/8 into LDS, track per-row max ---------------
  const _Float16* arow = Qh + ((size_t)bh * S_LEN + m0 + l) * HDIM;
  const v16h a0 = a_frag_f16(arow, 16 * half);
  const v16h a1 = a_frag_f16(arow, 32 + 16 * half);
  float lmax[8];
#pragma unroll
  for (int i = 0; i < 8; ++i) lmax[i] = -__builtin_inff();

  for (int tt = 0; tt < 32; ++tt) {
    const int n0 = (tt * 8 + w) * 16;
    const _Float16* brow = Kh + ((size_t)bh * S_LEN + n0 + l) * HDIM;
    v8f acc = {};
    acc = wmma16(a0, b_frag_f16(brow, 16 * half), acc);
    acc = wmma16(a1, b_frag_f16(brow, 32 + 16 * half), acc);
#pragma unroll
    for (int i = 0; i < 8; ++i) {
      const float s = acc[i] * 0.125f;
      lmax[i] = fmaxf(lmax[i], s);
      sc[(i + 8 * half) * SC_STRIDE + n0 + l] = (_Float16)s;
    }
  }
#pragma unroll
  for (int i = 0; i < 8; ++i) scr8[w][i + 8 * half][l] = lmax[i];
  __syncthreads();

  // ---- phase 2: row max across waves/lanes --------------------------------
  {
    const int row = t >> 4, j = t & 15;
    float m = scr8[0][row][j];
#pragma unroll
    for (int ww = 1; ww < 8; ++ww) m = fmaxf(m, scr8[ww][row][j]);
    red[row][j] = m;
  }
  __syncthreads();
  if (t < 16) {
    float m = red[t][0];
#pragma unroll
    for (int j = 1; j < 16; ++j) m = fmaxf(m, red[t][j]);
    rowmax[t] = m;
  }
  __syncthreads();

  // ---- phase 3: exp in place + row sum ------------------------------------
  {
    const int row = t >> 4, j = t & 15;
    const float mx = rowmax[row];
    _Float16* srow = sc + row * SC_STRIDE;
    float sum = 0.f;
#pragma unroll 4
    for (int k = 0; k < 256; ++k) {
      const int c = j + (k << 4);
      const float e = __expf((float)srow[c] - mx);
      srow[c] = (_Float16)e;
      sum += e;
    }
    red[row][j] = sum;
  }
  __syncthreads();
  if (t < 16) {
    float s = 0.f;
#pragma unroll
    for (int j = 0; j < 16; ++j) s += red[t][j];
    rowinv[t] = 1.f / s;
  }
  __syncthreads();

  // ---- phase 4: the one compulsory HBM write: normalized probs ------------
  {
    const int row = t >> 4, j = t & 15;
    const float inv = rowinv[row];
    const _Float16* srow = sc + row * SC_STRIDE;
    float* orow = attn + ((size_t)bh * S_LEN + m0 + row) * S_LEN;
#pragma unroll 4
    for (int k = 0; k < 256; ++k) {
      const int c = j + (k << 4);
      orow[c] = (float)srow[c] * inv;
    }
  }

  // ---- phase 5: context = P @ V straight from LDS -------------------------
  {
    const int tile = w >> 1, khalf = w & 1;          // 2 waves per n-tile
    const int kbase = khalf * 2048;
    const _Float16* aLds = sc + l * SC_STRIDE + kbase;          // local row l
    const _Float16* brow =
        Vt + ((size_t)bh * HDIM + tile * 16 + l) * S_LEN + kbase;
    v8f acc = {};
    for (int kk = 0; kk < 2048; kk += 32)
      acc = wmma16(a_frag_f16(aLds, kk + 16 * half),
                   b_frag_f16(brow, kk + 16 * half), acc);
#pragma unroll
    for (int i = 0; i < 8; ++i) scr8[w][i + 8 * half][l] = acc[i];
  }
  __syncthreads();

  // combine the two K-halves, fold in 1/rowsum, store context f16
  {
    const int b_ = bh >> 3, h = bh & (NHEADS - 1);
#pragma unroll
    for (int rep = 0; rep < 4; ++rep) {
      const int idx = t + 256 * rep;                 // 4 tiles x 256 elements
      const int tile = idx >> 8, el = idx & 255, r = el >> 4, c = el & 15;
      const float v = (scr8[2 * tile][r][c] + scr8[2 * tile + 1][r][c]) *
                      rowinv[r];
      ctx[((size_t)b_ * S_LEN + m0 + r) * DMODEL + h * HDIM + tile * 16 + c] =
          (_Float16)v;
    }
  }
}

// ---------------------------------------------------------------------------
// 3) output = ctx @ W_o^T + b_o  (f32 into d_out[0 .. B*S*512))
// ---------------------------------------------------------------------------
__global__ void out_proj_kernel(const _Float16* __restrict__ ctx,
                                const float* __restrict__ Wo,
                                const float* __restrict__ bo,
                                float* __restrict__ out) {
  const int lane = threadIdx.x, l = lane & 15, half = lane >> 4;
  const int m0 = blockIdx.y * 16, n0 = blockIdx.x * 16;
  const _Float16* arow = ctx + (size_t)(m0 + l) * DMODEL;
  const float*    brow = Wo  + (size_t)(n0 + l) * DMODEL;
  v8f acc = {};
#pragma unroll 4
  for (int kk = 0; kk < DMODEL; kk += 32)
    acc = wmma16(a_frag_f16(arow, kk + 16 * half),
                 b_frag_f32(brow, kk + 16 * half), acc);
  const float bv = bo[n0 + l];
#pragma unroll
  for (int i = 0; i < 8; ++i) {
    const int m = m0 + i + 8 * half;
    out[(size_t)m * DMODEL + n0 + l] = acc[i] + bv;
  }
}

// ---------------------------------------------------------------------------
extern "C" void kernel_launch(void* const* d_in, const int* in_sizes, int n_in,
                              void* d_out, int out_size, void* d_ws, size_t ws_size,
                              hipStream_t stream) {
  const float* query = (const float*)d_in[0];
  const float* key_  = (const float*)d_in[1];
  const float* value = (const float*)d_in[2];
  const float* w_q   = (const float*)d_in[3];
  const float* b_q   = (const float*)d_in[4];
  const float* w_k   = (const float*)d_in[5];
  const float* b_k   = (const float*)d_in[6];
  const float* w_v   = (const float*)d_in[7];
  const float* b_v   = (const float*)d_in[8];
  const float* w_o   = (const float*)d_in[9];
  const float* b_o   = (const float*)d_in[10];

  float* out0 = (float*)d_out;                                   // [2,4096,512]
  float* attn = out0 + (size_t)BATCH * S_LEN * DMODEL;           // [2,8,4096,4096]

  const size_t nel = (size_t)BATCH * NHEADS * S_LEN * HDIM;      // 4 Mi f16 each
  _Float16* Qh  = (_Float16*)d_ws;
  _Float16* Kh  = Qh + nel;
  _Float16* Vt  = Kh + nel;
  _Float16* ctx = Vt + nel;

  const dim3 projGrid(DMODEL / 16, (BATCH * S_LEN) / 16 / 8);    // 32 x 64 blocks
  qkv_proj_kernel<<<projGrid, 256, 0, stream>>>(query, w_q, b_q, Qh, 0);
  qkv_proj_kernel<<<projGrid, 256, 0, stream>>>(key_,  w_k, b_k, Kh, 0);
  qkv_proj_kernel<<<projGrid, 256, 0, stream>>>(value, w_v, b_v, Vt, 1);

  attn_fused_kernel<<<dim3(S_LEN / 16, BATCH * NHEADS), 256, 0, stream>>>(
      Qh, Kh, Vt, attn, ctx);

  out_proj_kernel<<<dim3(DMODEL / 16, (BATCH * S_LEN) / 16), 32, 0, stream>>>(
      ctx, w_o, b_o, out0);
}
